// CausalAttention_74869869903917
// MI455X (gfx1250) — matrix operand order repaired and checked by reading, not compiled, not run on hardware
//
#include <hip/hip_runtime.h>

// CDNA5 / gfx1250 causal-attention pipeline (wave32, WMMA bf16->f32,
// async global->LDS double buffering, ALiBi + causal flash attention).

typedef __bf16 bf16_t;
typedef __attribute__((ext_vector_type(16))) __bf16 v16bf;
typedef __attribute__((ext_vector_type(4)))  __bf16 v4bf;
typedef __attribute__((ext_vector_type(8)))  float  v8f;

constexpr int Bb = 2, Tt = 2048, Dd = 1024, Hh = 16, HD = 64;
constexpr int Mrows = Bb * Tt;  // 4096

__device__ __forceinline__ v8f wmma_bf16(v16bf a, v16bf b, v8f c) {
  return __builtin_amdgcn_wmma_f32_16x16x32_bf16(false, a, false, b, (short)0, c,
                                                 false, false);
}

// Low 32 bits of a generic pointer to LDS == LDS byte offset (ISA 10.2).
__device__ __forceinline__ unsigned lds_addr_of(const void* p) {
  return (unsigned)(uintptr_t)p;
}

// Async DMA: 16 bytes global -> LDS per lane, tracked by ASYNCcnt.
__device__ __forceinline__ void async_copy_b128(unsigned lds_byte_addr,
                                                const void* gaddr) {
  asm volatile("global_load_async_to_lds_b128 %0, %1, off"
               :
               : "v"(lds_byte_addr), "v"(gaddr)
               : "memory");
}

__device__ __forceinline__ void wait_async0() {
  asm volatile("s_wait_asynccnt 0" ::: "memory");
}

// ---------------------------------------------------------------------------
// One-time fp32 -> bf16 conversion (vectorized, memory-bound).
// ---------------------------------------------------------------------------
__global__ __launch_bounds__(256) void cvt_f32_bf16_kernel(
    const float* __restrict__ src, bf16_t* __restrict__ dst, int n4) {
  const int i = blockIdx.x * 256 + threadIdx.x;
  if (i < n4) {
    const float4 v = ((const float4*)src)[i];
    v4bf o = {(bf16_t)v.x, (bf16_t)v.y, (bf16_t)v.z, (bf16_t)v.w};
    ((v4bf*)dst)[i] = o;
  }
}

// ---------------------------------------------------------------------------
// One-time fp32 [1024][1024] -> bf16 transposed [1024][1024] (for Wq/Wk/Wv/Wo)
// so GEMM B-fragments become contiguous LDS reads.
// ---------------------------------------------------------------------------
__global__ __launch_bounds__(256) void cvt_transpose_kernel(
    const float* __restrict__ src, bf16_t* __restrict__ dst) {
  __shared__ float tile[32][33];
  const int bx = blockIdx.x * 32, by = blockIdx.y * 32;
  const int tx = threadIdx.x & 31, ty = threadIdx.x >> 5;  // 32 x 8
#pragma unroll
  for (int i = 0; i < 32; i += 8)
    tile[ty + i][tx] = src[(size_t)(by + ty + i) * Dd + bx + tx];
  __syncthreads();
#pragma unroll
  for (int i = 0; i < 32; i += 8)
    dst[(size_t)(bx + ty + i) * Dd + by + tx] = (bf16_t)tile[tx][ty + i];
}

// ---------------------------------------------------------------------------
// Tiled GEMM: C[M,N] = A[M,1024] @ WT[N,1024]^T + bias (A, WT bf16 in global).
// Block: 256 threads (8 waves, 4(M) x 2(N)); tile 128 x 64; K-step 32.
// Async global->LDS staging, double buffered, one barrier per K-step.
// LAYOUT 0: fp32 row-major out. 1: bf16 [B,H,T,64] (q/k). 2: bf16 [B,H,64,T] (v)
// ---------------------------------------------------------------------------
template <int LAYOUT>
__global__ __launch_bounds__(256) void gemm_bias_kernel(
    const bf16_t* __restrict__ A, const bf16_t* __restrict__ WT,
    const float* __restrict__ bias, void* __restrict__ Cout, float scale) {
  // padded row strides: 40 elements = 80 B (16B-aligned chunks, bank-spread)
  __shared__ __align__(16) bf16_t ldsA[2][128 * 40];
  __shared__ __align__(16) bf16_t ldsWt[2][64 * 40];

  const int tid  = threadIdx.x;
  const int lane = tid & 31;
  const int wv   = tid >> 5;
  const int hh   = lane >> 4;
  const int ln   = lane & 15;
  const int wm   = wv & 3;
  const int wn   = wv >> 2;
  const int row0 = blockIdx.x * 128;
  const int col0 = blockIdx.y * 64;

  const unsigned aLds = lds_addr_of(&ldsA[0][0]);
  const unsigned wLds = lds_addr_of(&ldsWt[0][0]);

  auto stageTiles = [&](int kk, int buf) {
    const unsigned aB = aLds + (unsigned)buf * (128u * 40u * 2u);
    const unsigned wB = wLds + (unsigned)buf * (64u * 40u * 2u);
    // A tile: 128 rows x 32 K = 512 x 16B chunks
#pragma unroll
    for (int i = 0; i < 2; ++i) {
      const int idx = tid + i * 256;
      const int r = idx >> 2, c = idx & 3;
      async_copy_b128(aB + (unsigned)(r * 80 + c * 16),
                      (const void*)(A + (size_t)(row0 + r) * Dd + kk + c * 8));
    }
    // W tile (transposed source): 64 rows(N) x 32 K = 256 x 16B chunks
    {
      const int r = tid >> 2, c = tid & 3;
      async_copy_b128(wB + (unsigned)(r * 80 + c * 16),
                      (const void*)(WT + (size_t)(col0 + r) * Dd + kk + c * 8));
    }
  };

  v8f zero = {};
  v8f acc[2][2];
#pragma unroll
  for (int i = 0; i < 2; ++i)
#pragma unroll
    for (int j = 0; j < 2; ++j) acc[i][j] = zero;

  stageTiles(0, 0);
  int cur = 0;
  for (int kk = 0; kk < Dd; kk += 32) {
    wait_async0();
    __syncthreads();
    if (kk + 32 < Dd) stageTiles(kk + 32, cur ^ 1);

    const bf16_t* At = ldsA[cur];
    const bf16_t* Wt = ldsWt[cur];
    v16bf aF[2], bF[2];
#pragma unroll
    for (int i = 0; i < 2; ++i) {
      const int m = 32 * wm + 16 * i + ln;
#pragma unroll
      for (int e = 0; e < 16; ++e) {
        const int K = e + ((e >= 8) ? 8 : 0) + (hh ? 8 : 0);  // A K striping
        aF[i][e] = At[m * 40 + K];
      }
    }
#pragma unroll
    for (int j = 0; j < 2; ++j) {
      const int n = 32 * wn + 16 * j + ln;
#pragma unroll
      for (int e = 0; e < 16; ++e) {
        const int K = e + ((e >= 8) ? 8 : 0) + (hh ? 8 : 0);  // B K striping
        bF[j][e] = Wt[n * 40 + K];
      }
    }
#pragma unroll
    for (int i = 0; i < 2; ++i)
#pragma unroll
      for (int j = 0; j < 2; ++j) acc[i][j] = wmma_bf16(aF[i], bF[j], acc[i][j]);
    cur ^= 1;
  }

  // epilogue: C element r -> (row = r + 8*hh, col = ln)
#pragma unroll
  for (int i = 0; i < 2; ++i) {
#pragma unroll
    for (int j = 0; j < 2; ++j) {
      const int col = col0 + 32 * wn + 16 * j + ln;
      const float bval = bias[col];
#pragma unroll
      for (int r = 0; r < 8; ++r) {
        const int row = row0 + 32 * wm + 16 * i + r + 8 * hh;
        const float v = (acc[i][j][r] + bval) * scale;
        if (LAYOUT == 0) {
          ((float*)Cout)[(size_t)row * Dd + col] = v;
        } else {
          const int bi = row >> 11, t = row & (Tt - 1);
          const int head = col >> 6, dd = col & 63;
          if (LAYOUT == 1)  // [B,H,T,hd] for q,k
            ((bf16_t*)Cout)[(((size_t)(bi * Hh + head)) * Tt + t) * HD + dd] =
                (bf16_t)v;
          else              // [B,H,hd,T] for v (transposed per head)
            ((bf16_t*)Cout)[(((size_t)(bi * Hh + head)) * HD + dd) * Tt + t] =
                (bf16_t)v;
        }
      }
    }
  }
}

// ---------------------------------------------------------------------------
// Flash attention with ALiBi + causal mask.
// Grid: (B*H, T/64). Block: 128 threads = 4 waves, each owns 16 query rows.
// q,k: bf16 [B,H,T,64] (q pre-scaled by 1/8); v: bf16 [B,H,64,T].
// Output: bf16 [B*T, D] for the final projection GEMM.
// ---------------------------------------------------------------------------
__global__ __launch_bounds__(128) void flash_attn_kernel(
    const bf16_t* __restrict__ qb, const bf16_t* __restrict__ kb,
    const bf16_t* __restrict__ vb, bf16_t* __restrict__ ab) {
  __shared__ __align__(16) bf16_t ldsK[2][32 * 72];   // [key][d], padded
  __shared__ __align__(16) bf16_t ldsVt[2][64 * 40];  // [d][key], padded
  __shared__ __align__(16) bf16_t ldsP[4][16 * 40];   // per-wave P relayout

  const int tid  = threadIdx.x;
  const int lane = tid & 31;
  const int wv   = tid >> 5;
  const int hh   = lane >> 4;
  const int ln   = lane & 15;
  const int bh   = blockIdx.x;
  const int q0   = blockIdx.y * 64;
  const int bi   = bh >> 4;
  const int hed  = bh & 15;
  const size_t headBase  = (size_t)bh * Tt * HD;  // q/k per-head base
  const size_t vHeadBase = (size_t)bh * HD * Tt;  // v per-head base

  const unsigned kLds = lds_addr_of(&ldsK[0][0]);
  const unsigned vLds = lds_addr_of(&ldsVt[0][0]);

  auto stageKV = [&](int j0, int buf) {
    const unsigned kB = kLds + (unsigned)buf * (32u * 72u * 2u);
    const unsigned vB = vLds + (unsigned)buf * (64u * 40u * 2u);
#pragma unroll
    for (int i = 0; i < 2; ++i) {
      const int idx = tid + i * 128;  // 0..255
      // K tile: 32 rows(key) x 64 d = 32 x 8 chunks
      const int kr = idx >> 3, kc = idx & 7;
      async_copy_b128(kB + (unsigned)(kr * 144 + kc * 16),
                      (const void*)(kb + headBase + (size_t)(j0 + kr) * HD +
                                    kc * 8));
      // V tile (transposed source): 64 rows(d) x 32 keys = 64 x 4 chunks
      const int vr = idx >> 2, vc = idx & 3;
      async_copy_b128(vB + (unsigned)(vr * 80 + vc * 16),
                      (const void*)(vb + vHeadBase + (size_t)vr * Tt + j0 +
                                    vc * 8));
    }
  };

  const float slope = exp2f(-0.5f * (float)(hed + 1));  // get_slopes(16)

  // resident Q fragments: 16 rows x 64 dims -> two K32 A-fragments
  v16bf aQ[2];
  const int qrow = q0 + 16 * wv + ln;
#pragma unroll
  for (int c = 0; c < 2; ++c) {
#pragma unroll
    for (int e = 0; e < 16; ++e) {
      const int kd = c * 32 + e + ((e >= 8) ? 8 : 0) + (hh ? 8 : 0);
      aQ[c][e] = qb[headBase + (size_t)qrow * HD + kd];
    }
  }

  float mstate[8], lstate[8];
#pragma unroll
  for (int r = 0; r < 8; ++r) { mstate[r] = -1e30f; lstate[r] = 0.0f; }
  v8f zero = {};
  v8f oacc[4];
#pragma unroll
  for (int dn = 0; dn < 4; ++dn) oacc[dn] = zero;

  stageKV(0, 0);
  int cur = 0;
  const int jEnd = q0 + 64;
  for (int j0 = 0; j0 < jEnd; j0 += 32) {
    wait_async0();
    __syncthreads();
    if (j0 + 32 < jEnd) stageKV(j0 + 32, cur ^ 1);
    if (j0 + 64 < jEnd) {  // prefetch tile-after-next toward L2
      __builtin_prefetch(kb + headBase + (size_t)(j0 + 64) * HD, 0, 3);
      __builtin_prefetch(vb + vHeadBase + (size_t)(j0 + 64), 0, 3);
    }

    const bf16_t* Kt = ldsK[cur];
    const bf16_t* Vt = ldsVt[cur];

    // S(16x32) = Q(16x64) . K^T
    v8f sAcc[2];
#pragma unroll
    for (int s = 0; s < 2; ++s) {
      v8f accS = zero;
#pragma unroll
      for (int c = 0; c < 2; ++c) {
        v16bf bK;
#pragma unroll
        for (int e = 0; e < 16; ++e) {
          const int kd = c * 32 + e + ((e >= 8) ? 8 : 0) + (hh ? 8 : 0);
          bK[e] = Kt[(16 * s + ln) * 72 + kd];  // lane = key column
        }
        accS = wmma_bf16(aQ[c], bK, accS);
      }
      sAcc[s] = accS;
    }

    // ALiBi (key-only) + causal mask on C fragments
#pragma unroll
    for (int s = 0; s < 2; ++s) {
      const int kidx = j0 + 16 * s + ln;
      const float al = -slope * (float)(Tt - 1 - kidx);
#pragma unroll
      for (int r = 0; r < 8; ++r) {
        const int qi = q0 + 16 * wv + r + 8 * hh;
        float v = sAcc[s][r] + al;
        if (kidx > qi) v = -1e30f;
        sAcc[s][r] = v;
      }
    }

    // online softmax (row reductions across 16 lanes of each half)
    float pvv[2][8];
#pragma unroll
    for (int r = 0; r < 8; ++r) {
      float mx = fmaxf(sAcc[0][r], sAcc[1][r]);
      mx = fmaxf(mx, __shfl_xor(mx, 8, 32));
      mx = fmaxf(mx, __shfl_xor(mx, 4, 32));
      mx = fmaxf(mx, __shfl_xor(mx, 2, 32));
      mx = fmaxf(mx, __shfl_xor(mx, 1, 32));
      const float mnew = fmaxf(mstate[r], mx);
      const float corr = __expf(mstate[r] - mnew);
      mstate[r] = mnew;
      const float p0 = __expf(sAcc[0][r] - mnew);
      const float p1 = __expf(sAcc[1][r] - mnew);
      pvv[0][r] = p0;
      pvv[1][r] = p1;
      float rs = p0 + p1;
      rs += __shfl_xor(rs, 8, 32);
      rs += __shfl_xor(rs, 4, 32);
      rs += __shfl_xor(rs, 2, 32);
      rs += __shfl_xor(rs, 1, 32);
      lstate[r] = lstate[r] * corr + rs;
#pragma unroll
      for (int dn = 0; dn < 4; ++dn) oacc[dn][r] *= corr;
    }

    // P: C layout -> per-wave LDS [16 x 32] (wave-private, LDS is in-order)
#pragma unroll
    for (int s = 0; s < 2; ++s)
#pragma unroll
      for (int r = 0; r < 8; ++r)
        ldsP[wv][(r + 8 * hh) * 40 + 16 * s + ln] = (bf16_t)pvv[s][r];

    v16bf pF;
#pragma unroll
    for (int e = 0; e < 16; ++e) {
      const int K = e + ((e >= 8) ? 8 : 0) + (hh ? 8 : 0);
      pF[e] = ldsP[wv][ln * 40 + K];
    }

    // O(16x64) += P(16x32) . V(32x64)  (V staged transposed: [d][key])
#pragma unroll
    for (int dn = 0; dn < 4; ++dn) {
      v16bf bV;
#pragma unroll
      for (int e = 0; e < 16; ++e) {
        const int key = e + ((e >= 8) ? 8 : 0) + (hh ? 8 : 0);
        bV[e] = Vt[(dn * 16 + ln) * 40 + key];
      }
      oacc[dn] = wmma_bf16(pF, bV, oacc[dn]);
    }
    cur ^= 1;
  }

  // epilogue: O / l -> bf16 [B*T, D]
#pragma unroll
  for (int r = 0; r < 8; ++r) {
    const float inv = 1.0f / lstate[r];
    const int t = q0 + 16 * wv + r + 8 * hh;
    const size_t rowBase = ((size_t)bi * Tt + t) * Dd + (size_t)hed * HD;
#pragma unroll
    for (int dn = 0; dn < 4; ++dn)
      ab[rowBase + dn * 16 + ln] = (bf16_t)(oacc[dn][r] * inv);
  }
}

// ---------------------------------------------------------------------------
extern "C" void kernel_launch(void* const* d_in, const int* in_sizes, int n_in,
                              void* d_out, int out_size, void* d_ws,
                              size_t ws_size, hipStream_t stream) {
  const float* x  = (const float*)d_in[0];
  const float* Wq = (const float*)d_in[1];
  const float* bq = (const float*)d_in[2];
  const float* Wk = (const float*)d_in[3];
  const float* bk = (const float*)d_in[4];
  const float* Wv = (const float*)d_in[5];
  const float* bv = (const float*)d_in[6];
  const float* Wo = (const float*)d_in[7];
  const float* bo = (const float*)d_in[8];

  const size_t XN = (size_t)Mrows * Dd;       // 4,194,304
  const size_t WN = (size_t)Dd * Dd;          // 1,048,576
  bf16_t* xb   = (bf16_t*)d_ws;
  bf16_t* wqT  = xb + XN;
  bf16_t* wkT  = wqT + WN;
  bf16_t* wvT  = wkT + WN;
  bf16_t* woT  = wvT + WN;
  bf16_t* qbuf = woT + WN;
  bf16_t* kbuf = qbuf + XN;
  bf16_t* vbuf = kbuf + XN;
  bf16_t* abuf = vbuf + XN;

  // one-time conversions
  cvt_f32_bf16_kernel<<<(int)(XN / 4 / 256), 256, 0, stream>>>(x, xb,
                                                               (int)(XN / 4));
  dim3 tg(Dd / 32, Dd / 32);
  cvt_transpose_kernel<<<tg, 256, 0, stream>>>(Wq, wqT);
  cvt_transpose_kernel<<<tg, 256, 0, stream>>>(Wk, wkT);
  cvt_transpose_kernel<<<tg, 256, 0, stream>>>(Wv, wvT);
  cvt_transpose_kernel<<<tg, 256, 0, stream>>>(Wo, woT);

  // QKV projections (q folded with 1/sqrt(64))
  dim3 gg(Mrows / 128, Dd / 64);
  gemm_bias_kernel<1><<<gg, 256, 0, stream>>>(xb, wqT, bq, qbuf, 0.125f);
  gemm_bias_kernel<1><<<gg, 256, 0, stream>>>(xb, wkT, bk, kbuf, 1.0f);
  gemm_bias_kernel<2><<<gg, 256, 0, stream>>>(xb, wvT, bv, vbuf, 1.0f);

  flash_attn_kernel<<<dim3(Bb * Hh, Tt / 64), 128, 0, stream>>>(qbuf, kbuf,
                                                                vbuf, abuf);

  gemm_bias_kernel<0><<<gg, 256, 0, stream>>>(abuf, woT, bo, d_out, 1.0f);
}
